// MultiFrameFeatsFusionBlock_55637006353066
// MI455X (gfx1250) — matrix operand order repaired and checked by reading, not compiled, not run on hardware
//
#include <hip/hip_runtime.h>

// MultiFrameFeatsFusionBlock for MI455X (gfx1250, wave32, WMMA).
// B=2, C=64, H=W=96, N=9216.
//
// Pipeline:
//  k_conv   : theta/phi 1x1 convs -> xh[b][n][c] (f16), yh[b][m][c] (f16)
//  k_colsum : dinv[b][m] = 1 / sum_n exp(S[n,m])   (S via v_wmma f16, f32 acc)
//  k_xsT    : xsT[b][c][m] = bf16( xh[b][m][c] * dinv[b][m] )  (LDS transpose)
//  k_wh     : Wwh = f16(W_w)
//  k_attn   : per 16-row n-tile: stream m in chunks of 32:
//               S^T tiles via wmma_f16, E=exp(S) (bf16, f32-range safe),
//               z^T += xsT x E via wmma_bf16 (E relayout via shfl_xor only),
//             then fused final 1x1 conv via wmma_f16 and coalesced f32 store.

#define Bn 2
#define Cc 64
#define Nn 9216
#define NT 576      // Nn/16
#define MC 288      // Nn/32

typedef _Float16 v16h  __attribute__((ext_vector_type(16)));
typedef __bf16   v16bf __attribute__((ext_vector_type(16)));
typedef float    v8f   __attribute__((ext_vector_type(8)));
typedef float    f32x4 __attribute__((ext_vector_type(4)));

union U16h { v16h  m; f32x4 q[2]; };
union U16b { v16bf m; f32x4 q[2]; };

// ---------------- kernel 1: theta & phi 1x1 convs, output [n][c] f16 ----------------
__global__ __launch_bounds__(256) void k_conv(const float* __restrict__ frame,
                                              const float* __restrict__ thw,
                                              const float* __restrict__ thb,
                                              const float* __restrict__ phw,
                                              const float* __restrict__ phb,
                                              _Float16* __restrict__ xh,
                                              _Float16* __restrict__ yh) {
  __shared__ float fr[64][64];   // frame tile [k][n_local]
  __shared__ float wt[64][65];   // theta_w [o][k] (padded: no bank conflicts)
  __shared__ float wp[64][65];   // phi_w   [o][k]
  const int t  = threadIdx.x;
  const int b  = blockIdx.x / (Nn / 64);
  const int n0 = (blockIdx.x % (Nn / 64)) * 64;
  for (int i = 0; i < 16; ++i) {
    int idx = t + 256 * i;
    int r = idx >> 6, c = idx & 63;
    fr[r][c] = frame[(size_t)(b * Cc + r) * Nn + n0 + c];
    wt[r][c] = thw[idx];
    wp[r][c] = phw[idx];
  }
  __syncthreads();
  const int o   = t & 63;
  const int nlb = t >> 6;          // 0..3; wave-uniform
  const float bT = thb[o], bP = phb[o];
  float aT[16], aP[16];
#pragma unroll
  for (int i = 0; i < 16; ++i) { aT[i] = bT; aP[i] = bP; }
  for (int k = 0; k < 64; ++k) {
    float w0 = wt[o][k], w1 = wp[o][k];
#pragma unroll
    for (int i = 0; i < 16; ++i) {
      float f = fr[k][nlb + 4 * i];   // LDS broadcast across lanes
      aT[i] += w0 * f;
      aP[i] += w1 * f;
    }
  }
#pragma unroll
  for (int i = 0; i < 16; ++i) {
    int n = n0 + nlb + 4 * i;
    xh[(size_t)(b * Nn + n) * 64 + o] = (_Float16)aT[i];
    yh[(size_t)(b * Nn + n) * 64 + o] = (_Float16)aP[i];
  }
}

// ---------------- kernel 2: colsum pass, 1 wave per 16 columns of m ----------------
__global__ __launch_bounds__(256) void k_colsum(const _Float16* __restrict__ xh,
                                                const _Float16* __restrict__ yh,
                                                float* __restrict__ dinv) {
  const int lane = threadIdx.x & 31;
  const int flat = blockIdx.x * 8 + (threadIdx.x >> 5);
  const int b    = flat / NT;
  const int m0   = (flat % NT) * 16;
  const int lr = lane & 15, lh = lane >> 4;

  // B operand: y columns m0..m0+15, K = channel (two k-steps of 32)
  const _Float16* yrow = yh + (size_t)(b * Nn + m0 + lr) * 64;
  U16h By0, By1;
  By0.q[0] = *(const f32x4*)(yrow + lh * 8);
  By0.q[1] = *(const f32x4*)(yrow + 16 + lh * 8);
  By1.q[0] = *(const f32x4*)(yrow + 32 + lh * 8);
  By1.q[1] = *(const f32x4*)(yrow + 48 + lh * 8);

  float rsum = 0.0f;
  for (int nt = 0; nt < NT; ++nt) {
    // A operand: x rows nt*16..+15, K split 8+8 per lane-half
    const _Float16* xrow = xh + (size_t)(b * Nn + nt * 16 + lr) * 64;
    U16h A0, A1;
    A0.q[0] = *(const f32x4*)(xrow + lh * 8);
    A0.q[1] = *(const f32x4*)(xrow + 16 + lh * 8);
    A1.q[0] = *(const f32x4*)(xrow + 32 + lh * 8);
    A1.q[1] = *(const f32x4*)(xrow + 48 + lh * 8);
    v8f S = {0.f, 0.f, 0.f, 0.f, 0.f, 0.f, 0.f, 0.f};
    S = __builtin_amdgcn_wmma_f32_16x16x32_f16(false, A0.m, false, By0.m, (short)0, S, false, false);
    S = __builtin_amdgcn_wmma_f32_16x16x32_f16(false, A1.m, false, By1.m, (short)0, S, false, false);
#pragma unroll
    for (int v = 0; v < 8; ++v) rsum += __expf(S[v]);
  }
  float tot = rsum + __shfl_xor(rsum, 16, 32);
  if (lane < 16) dinv[b * Nn + m0 + lane] = 1.0f / tot;
}

// ---------------- kernel 3a: xsT[b][c][m] = bf16(xh[b][m][c] * dinv[m]) ----------------
__global__ __launch_bounds__(256) void k_xsT(const _Float16* __restrict__ xh,
                                             const float* __restrict__ dinv,
                                             __bf16* __restrict__ xsT) {
  __shared__ __bf16 tile[64][65];
  const int t  = threadIdx.x;
  const int b  = blockIdx.x / (Nn / 64);
  const int m0 = (blockIdx.x % (Nn / 64)) * 64;
  const int c = t & 63, mb = t >> 6;
#pragma unroll
  for (int j = 0; j < 16; ++j) {
    int ml = mb + 4 * j;
    float v = (float)xh[(size_t)(b * Nn + m0 + ml) * 64 + c] * dinv[b * Nn + m0 + ml];
    tile[c][ml] = (__bf16)v;
  }
  __syncthreads();
  const int ml2 = t & 63, cb = t >> 6;
#pragma unroll
  for (int j = 0; j < 16; ++j) {
    int cl = cb + 4 * j;
    xsT[(size_t)(b * Cc + cl) * Nn + m0 + ml2] = tile[cl][ml2];
  }
}

// ---------------- kernel 3b: W_w -> f16 ----------------
__global__ __launch_bounds__(256) void k_wh(const float* __restrict__ Ww,
                                            _Float16* __restrict__ Wwh) {
  const int t = threadIdx.x;
#pragma unroll
  for (int i = 0; i < 16; ++i) Wwh[t + 256 * i] = (_Float16)Ww[t + 256 * i];
}

// ---------------- kernel 4: streaming attention + fused final conv ----------------
__global__ __launch_bounds__(256) void k_attn(const _Float16* __restrict__ xh,
                                              const _Float16* __restrict__ yh,
                                              const __bf16* __restrict__ xsT,
                                              const _Float16* __restrict__ Wwh,
                                              const float* __restrict__ Wb,
                                              float* __restrict__ out) {
  const int lane = threadIdx.x & 31;
  const int flat = blockIdx.x * 8 + (threadIdx.x >> 5);
  const int b    = flat / NT;
  const int n0   = (flat % NT) * 16;
  const int lr = lane & 15, lh = lane >> 4;
  const bool low = lane < 16;

  // B operand of the S^T GEMM: x columns n0..n0+15, K = channel (fixed all loop)
  const _Float16* xrow = xh + (size_t)(b * Nn + n0 + lr) * 64;
  U16h Bx0, Bx1;
  Bx0.q[0] = *(const f32x4*)(xrow + lh * 8);
  Bx0.q[1] = *(const f32x4*)(xrow + 16 + lh * 8);
  Bx1.q[0] = *(const f32x4*)(xrow + 32 + lh * 8);
  Bx1.q[1] = *(const f32x4*)(xrow + 48 + lh * 8);

  v8f z[4];
#pragma unroll
  for (int ct = 0; ct < 4; ++ct) z[ct] = (v8f){0.f, 0.f, 0.f, 0.f, 0.f, 0.f, 0.f, 0.f};

  for (int mc = 0; mc < MC; ++mc) {
    const int m0 = mc * 32;
    // A operands: y rows (two 16-row sub-tiles), two k-steps over channels
    const _Float16* y0 = yh + (size_t)(b * Nn + m0 + lr) * 64;
    const _Float16* y1 = yh + (size_t)(b * Nn + m0 + 16 + lr) * 64;
    U16h A00, A01, A10, A11;
    A00.q[0] = *(const f32x4*)(y0 + lh * 8);       A00.q[1] = *(const f32x4*)(y0 + 16 + lh * 8);
    A01.q[0] = *(const f32x4*)(y0 + 32 + lh * 8);  A01.q[1] = *(const f32x4*)(y0 + 48 + lh * 8);
    A10.q[0] = *(const f32x4*)(y1 + lh * 8);       A10.q[1] = *(const f32x4*)(y1 + 16 + lh * 8);
    A11.q[0] = *(const f32x4*)(y1 + 32 + lh * 8);  A11.q[1] = *(const f32x4*)(y1 + 48 + lh * 8);

    v8f S0 = {0.f, 0.f, 0.f, 0.f, 0.f, 0.f, 0.f, 0.f};
    v8f S1 = {0.f, 0.f, 0.f, 0.f, 0.f, 0.f, 0.f, 0.f};
    S0 = __builtin_amdgcn_wmma_f32_16x16x32_f16(false, A00.m, false, Bx0.m, (short)0, S0, false, false);
    S0 = __builtin_amdgcn_wmma_f32_16x16x32_f16(false, A01.m, false, Bx1.m, (short)0, S0, false, false);
    S1 = __builtin_amdgcn_wmma_f32_16x16x32_f16(false, A10.m, false, Bx0.m, (short)0, S1, false, false);
    S1 = __builtin_amdgcn_wmma_f32_16x16x32_f16(false, A11.m, false, Bx1.m, (short)0, S1, false, false);

    float e0[8], e1[8];
#pragma unroll
    for (int v = 0; v < 8; ++v) { e0[v] = __expf(S0[v]); e1[v] = __expf(S1[v]); }

    // Relayout exp(S^T) D-tiles into the B operand (K=32 m-values, cols n):
    // only a lane-half swap is needed -> 16 shfl_xor, no LDS.
    v16bf Bz;
#pragma unroll
    for (int j = 0; j < 8; ++j) {
      float o0 = __shfl_xor(e0[j], 16, 32);
      float o1 = __shfl_xor(e1[j], 16, 32);
      Bz[j]     = (__bf16)(low ? e0[j] : o1);
      Bz[j + 8] = (__bf16)(low ? o0 : e1[j]);
    }

    // z^T[c][n] += xsT (A: rows c, K=m) x E (B) ; bf16 inputs, f32 accum
#pragma unroll
    for (int ct = 0; ct < 4; ++ct) {
      const __bf16* xr = xsT + (size_t)(b * Cc + ct * 16 + lr) * Nn + m0;
      U16b Ax;
      Ax.q[0] = *(const f32x4*)(xr + lh * 16);
      Ax.q[1] = *(const f32x4*)(xr + lh * 16 + 8);
      z[ct] = __builtin_amdgcn_wmma_f32_16x16x32_bf16(false, Ax.m, false, Bz, (short)0, z[ct], false, false);
    }
  }

  // Fused final conv: out^T[o][n] = Ww (A, K=c) x z^T (B) + bias
  v16h Bz0, Bz1;
#pragma unroll
  for (int j = 0; j < 8; ++j) {
    float a0 = __shfl_xor(z[0][j], 16, 32);
    float a1 = __shfl_xor(z[1][j], 16, 32);
    float a2 = __shfl_xor(z[2][j], 16, 32);
    float a3 = __shfl_xor(z[3][j], 16, 32);
    Bz0[j]     = (_Float16)(low ? z[0][j] : a1);
    Bz0[j + 8] = (_Float16)(low ? a0 : z[1][j]);
    Bz1[j]     = (_Float16)(low ? z[2][j] : a3);
    Bz1[j + 8] = (_Float16)(low ? a2 : z[3][j]);
  }
#pragma unroll
  for (int ot = 0; ot < 4; ++ot) {
    const _Float16* wr = Wwh + (ot * 16 + lr) * 64;
    U16h W0, W1;
    W0.q[0] = *(const f32x4*)(wr + lh * 16);       W0.q[1] = *(const f32x4*)(wr + lh * 16 + 8);
    W1.q[0] = *(const f32x4*)(wr + 32 + lh * 16);  W1.q[1] = *(const f32x4*)(wr + 32 + lh * 16 + 8);
    v8f O = {0.f, 0.f, 0.f, 0.f, 0.f, 0.f, 0.f, 0.f};
    O = __builtin_amdgcn_wmma_f32_16x16x32_f16(false, W0.m, false, Bz0, (short)0, O, false, false);
    O = __builtin_amdgcn_wmma_f32_16x16x32_f16(false, W1.m, false, Bz1, (short)0, O, false, false);
#pragma unroll
    for (int v = 0; v < 8; ++v) {
      int o = ot * 16 + v + 8 * lh;
      out[(size_t)(b * Cc + o) * Nn + n0 + lr] = O[v] + Wb[o];
    }
  }
}

// ---------------- launch ----------------
extern "C" void kernel_launch(void* const* d_in, const int* in_sizes, int n_in,
                              void* d_out, int out_size, void* d_ws, size_t ws_size,
                              hipStream_t stream) {
  const float* frame = (const float*)d_in[0];
  const float* thw   = (const float*)d_in[1];
  const float* thb   = (const float*)d_in[2];
  const float* phw   = (const float*)d_in[3];
  const float* phb   = (const float*)d_in[4];
  const float* Ww    = (const float*)d_in[5];
  const float* Wb    = (const float*)d_in[6];
  float* out = (float*)d_out;

  char* ws = (char*)d_ws;
  _Float16* xh   = (_Float16*)(ws);                 // B*N*C f16 = 2,359,296 B
  _Float16* yh   = (_Float16*)(ws + 2359296);       // B*N*C f16
  __bf16*   xsT  = (__bf16*)  (ws + 4718592);       // B*C*N bf16
  _Float16* Wwh  = (_Float16*)(ws + 7077888);       // 64*64 f16 = 8192 B
  float*    dinv = (float*)   (ws + 7086080);       // B*N f32 = 73,728 B

  hipLaunchKernelGGL(k_conv,   dim3(Bn * (Nn / 64)), dim3(256), 0, stream,
                     frame, thw, thb, phw, phb, xh, yh);
  hipLaunchKernelGGL(k_colsum, dim3(Bn * NT / 8),    dim3(256), 0, stream, xh, yh, dinv);
  hipLaunchKernelGGL(k_xsT,    dim3(Bn * (Nn / 64)), dim3(256), 0, stream, xh, dinv, xsT);
  hipLaunchKernelGGL(k_wh,     dim3(1),              dim3(256), 0, stream, Ww, Wwh);
  hipLaunchKernelGGL(k_attn,   dim3(Bn * NT / 8),    dim3(256), 0, stream,
                     xh, yh, xsT, Wwh, Wb, out);
}
